// Model_90529320665777
// MI455X (gfx1250) — compile-verified, matrix operand order checked by reading, not run on hardware
//
#include <hip/hip_runtime.h>
#include <math.h>

typedef __attribute__((ext_vector_type(16))) _Float16 v16h;
typedef __attribute__((ext_vector_type(8)))  float    v8f;

#define B_  128
#define C_  256
#define F_  7
#define T_  251
#define NI  (B_*F_)      // 896
#define NBT (B_*T_)      // 32128
#define NB4 126          // ceil(NBT/256)

// ---- workspace layout (byte offsets, d_ws is 256B aligned) ----
static constexpr size_t O_AQC  = 0;
static constexpr size_t O_AKC  = 1024;
static constexpr size_t O_AQF  = 2048;
static constexpr size_t O_AKF  = 3072;
static constexpr size_t O_AQT  = 4096;
static constexpr size_t O_AKT  = 5120;
static constexpr size_t O_CONS = 6144;   // [0..5]=cq_c,ck_c,cq_f,ck_f,cq_t,ck_t  [6..8]=sc_c,sc_f,sc_t
static constexpr size_t O_SV   = 7168;
static constexpr size_t O_BETA = 8192;
static constexpr size_t O_DC   = 9216;
static constexpr size_t O_DF   = 10240;
static constexpr size_t O_DT   = 11264;
static constexpr size_t O_WVH  = 16384;                  // 256*256 f16 = 131072B
static constexpr size_t O_QPC  = O_WVH + 131072;         // 896*256 f32
static constexpr size_t O_KPC  = O_QPC + 917504;
static constexpr size_t O_QPF  = O_KPC + 917504;         // 896*251 f32 (padded)
static constexpr size_t O_KPF  = O_QPF + 901120;
static constexpr size_t O_QPT  = O_KPF + 901120;
static constexpr size_t O_KPT  = O_QPT + 901120;
static constexpr size_t O_CC   = O_KPT + 901120;         // contrib_c 896*256
static constexpr size_t O_CF   = O_CC  + 917504;         // contrib_f 896*251
static constexpr size_t O_PT   = O_CF  + 901120;         // partial_t NB4*7

__device__ __forceinline__ float bnscale() { return 1.0f / sqrtf(1.00001f); }

// ---- K0a: channel-branch coefficient vectors (length T) ----
__global__ void k0a(const float* __restrict__ wq, const float* __restrict__ wk,
                    const float* __restrict__ gq, const float* __restrict__ gk,
                    float* __restrict__ aq, float* __restrict__ ak) {
  int t = threadIdx.x;
  if (t >= T_) return;
  float s = bnscale() / (float)T_;
  float sq = 0.f, sk = 0.f;
  for (int o = 0; o < T_; ++o) { sq += wq[o*T_ + t]*gq[o]; sk += wk[o*T_ + t]*gk[o]; }
  aq[t] = sq*s; ak[t] = sk*s;
}

// ---- K0b: freq/time-branch coefficient vectors (length C) ----
__global__ void k0b(const float* __restrict__ wqf, const float* __restrict__ wkf,
                    const float* __restrict__ wqt, const float* __restrict__ wkt,
                    const float* __restrict__ gqf, const float* __restrict__ gkf,
                    const float* __restrict__ gqt, const float* __restrict__ gkt,
                    float* __restrict__ aqf, float* __restrict__ akf,
                    float* __restrict__ aqt, float* __restrict__ akt) {
  int c = threadIdx.x;
  float s = bnscale() / (float)C_;
  float s0=0.f, s1=0.f, s2=0.f, s3=0.f;
  for (int o = 0; o < C_; ++o) {
    s0 += wqf[o*C_+c]*gqf[o]; s1 += wkf[o*C_+c]*gkf[o];
    s2 += wqt[o*C_+c]*gqt[o]; s3 += wkt[o*C_+c]*gkt[o];
  }
  aqf[c]=s0*s; akf[c]=s1*s; aqt[c]=s2*s; akt[c]=s3*s;
}

// ---- K0c: wv -> f16 ----
__global__ void k0c(const float* __restrict__ wv, _Float16* __restrict__ wvh) {
  int idx = blockIdx.x*256 + threadIdx.x;
  wvh[idx] = (_Float16)wv[idx];
}

// ---- K0d: constants, sv/beta ----
__global__ void k0d(const float* bq_c, const float* bk_c, const float* bq_f, const float* bk_f,
                    const float* bq_t, const float* bk_t, const float* glog_c, const float* glog_f,
                    const float* glog_t, const float* gv, const float* bv, const float* bbv,
                    float* cons, float* sv, float* beta) {
  int tid = threadIdx.x;
  float s = bnscale();
  float svv = gv[tid]*s;
  sv[tid] = svv;
  beta[tid] = bv[tid]*svv + bbv[tid];
  if (tid == 0) { float a=0; for (int o=0;o<T_;++o) a+=bq_c[o]; cons[0]=a/(float)T_; }
  if (tid == 1) { float a=0; for (int o=0;o<T_;++o) a+=bk_c[o]; cons[1]=a/(float)T_; }
  if (tid == 2) { float a=0; for (int o=0;o<C_;++o) a+=bq_f[o]; cons[2]=a/(float)C_; }
  if (tid == 3) { float a=0; for (int o=0;o<C_;++o) a+=bk_f[o]; cons[3]=a/(float)C_; }
  if (tid == 4) { float a=0; for (int o=0;o<C_;++o) a+=bq_t[o]; cons[4]=a/(float)C_; }
  if (tid == 5) { cons[6]=glog_c[0]*s; cons[7]=glog_f[0]*s; cons[8]=glog_t[0]*s; }
  if (tid == 6) { float a=0; for (int o=0;o<C_;++o) a+=bk_t[o]; cons[5]=a/(float)C_; }
}

// ---- K1a: qp_c/kp_c  (one wave per (i,c) row; reduce over t) ----
__global__ void k1a(const float* __restrict__ x, const float* __restrict__ aq,
                    const float* __restrict__ ak, const float* __restrict__ cons,
                    float* __restrict__ qp, float* __restrict__ kp) {
  int row  = blockIdx.x*8 + (threadIdx.x >> 5);  // row = i*256 + c
  int lane = threadIdx.x & 31;
  int i = row >> 8, c = row & 255;
  int b = i / F_, f = i % F_;
  const float* xr = x + ((size_t)(b*C_ + c)*F_ + f)*T_;
  float sq = 0.f, sk = 0.f;
  for (int t = lane; t < T_; t += 32) {
    float xv = xr[t];
    sq += xv*aq[t]; sk += xv*ak[t];
  }
  for (int off = 16; off > 0; off >>= 1) {
    sq += __shfl_down(sq, off, 32);
    sk += __shfl_down(sk, off, 32);
  }
  if (lane == 0) { qp[row] = sq + cons[0]; kp[row] = sk + cons[1]; }
}

// ---- K1b: qp_f/kp_f/qp_t/kp_t  (one thread per (i,t); reduce over c) ----
__global__ void k1b(const float* __restrict__ x,
                    const float* __restrict__ aqf, const float* __restrict__ akf,
                    const float* __restrict__ aqt, const float* __restrict__ akt,
                    const float* __restrict__ cons,
                    float* __restrict__ qpf, float* __restrict__ kpf,
                    float* __restrict__ qpt, float* __restrict__ kpt) {
  int i = blockIdx.x, t = threadIdx.x;
  if (t >= T_) return;
  int b = i / F_, f = i % F_;
  const float* xb = x + ((size_t)b*C_*F_ + f)*T_ + t;
  float s0=0.f, s1=0.f, s2=0.f, s3=0.f;
  for (int c = 0; c < C_; ++c) {
    float xv = xb[(size_t)c*F_*T_];
    s0 += xv*aqf[c]; s1 += xv*akf[c]; s2 += xv*aqt[c]; s3 += xv*akt[c];
  }
  size_t o = (size_t)i*T_ + t;
  qpf[o]=s0+cons[2]; kpf[o]=s1+cons[3]; qpt[o]=s2+cons[4]; kpt[o]=s3+cons[5];
}

// ---- K2: channel-branch softmax diagonal contributions ----
__global__ void k2(const float* __restrict__ qp, const float* __restrict__ kp,
                   const float* __restrict__ cons, float* __restrict__ contrib) {
  __shared__ float kk[C_];
  int i = blockIdx.x, h = threadIdx.x;
  float s = cons[6];
  kk[h] = kp[(size_t)i*C_ + h];
  __syncthreads();
  float z = qp[(size_t)i*C_ + h] * s;
  float rs = 0.f;
  #pragma unroll 4
  for (int c = 0; c < C_; ++c) rs += __expf(z * kk[c]);
  contrib[(size_t)i*C_ + h] = __expf(z * kk[h]) / rs;
}

// ---- K3: frequency-branch softmax diagonal contributions ----
__global__ void k3(const float* __restrict__ qp, const float* __restrict__ kp,
                   const float* __restrict__ cons, float* __restrict__ contrib) {
  __shared__ float kk[T_];
  int i = blockIdx.x, t = threadIdx.x;
  float s = cons[7];
  if (t < T_) kk[t] = kp[(size_t)i*T_ + t];
  __syncthreads();
  if (t >= T_) return;
  float z = qp[(size_t)i*T_ + t] * s;
  float rs = 0.f;
  #pragma unroll 4
  for (int u = 0; u < T_; ++u) rs += __expf(z * kk[u]);
  contrib[(size_t)i*T_ + t] = __expf(z * kk[t]) / rs;
}

// ---- K4: time-branch (7x7 softmax per (b,t)), block-level fixed-order partials ----
__global__ void k4(const float* __restrict__ qp, const float* __restrict__ kp,
                   const float* __restrict__ cons, float* __restrict__ partial) {
  __shared__ float red[256][8];
  int idx = blockIdx.x*256 + threadIdx.x;
  float cf[7];
  #pragma unroll
  for (int f = 0; f < 7; ++f) cf[f] = 0.f;
  if (idx < NBT) {
    int b = idx / T_, t = idx % T_;
    float s = cons[8];
    float q[7], k[7];
    #pragma unroll
    for (int f = 0; f < 7; ++f) {
      size_t o = (size_t)(b*F_ + f)*T_ + t;
      q[f] = qp[o]*s; k[f] = kp[o];
    }
    #pragma unroll
    for (int f = 0; f < 7; ++f) {
      float rs = 0.f, e = 0.f;
      #pragma unroll
      for (int g = 0; g < 7; ++g) {
        float ex = __expf(q[f]*k[g]);
        rs += ex;
        if (g == f) e = ex;
      }
      cf[f] = e/rs;
    }
  }
  #pragma unroll
  for (int f = 0; f < 7; ++f) red[threadIdx.x][f] = cf[f];
  __syncthreads();
  if (threadIdx.x < 7) {
    float sum = 0.f;
    for (int r = 0; r < 256; ++r) sum += red[r][threadIdx.x];
    partial[blockIdx.x*7 + threadIdx.x] = sum;
  }
}

// ---- Kred: deterministic final reductions -> dc[256], df[251], dt[7] ----
__global__ void kred(const float* __restrict__ cc, const float* __restrict__ cf,
                     const float* __restrict__ pt,
                     float* __restrict__ dc, float* __restrict__ df, float* __restrict__ dt) {
  int tid = threadIdx.x;
  { float s=0.f; for (int i=0;i<NI;++i) s += cc[(size_t)i*C_ + tid]; dc[tid]=s; }
  if (tid < T_) { float s=0.f; for (int i=0;i<NI;++i) s += cf[(size_t)i*T_ + tid]; df[tid]=s; }
  if (tid < 7)  { float s=0.f; for (int b=0;b<NB4;++b) s += pt[b*7+tid]; dt[tid]=s; }
}

// ---- K5: fused WMMA matmul (v = wv@x) + scale + residual ----
// grid = 896*4 ; block = 256 (8 waves). Per block: M=256(h) x N=64(t) x K=256(c)
__global__ void k5(const float* __restrict__ x, const _Float16* __restrict__ wvh,
                   const float* __restrict__ sv, const float* __restrict__ beta,
                   const float* __restrict__ dc, const float* __restrict__ df,
                   const float* __restrict__ dt, float* __restrict__ out) {
  __shared__ __align__(32) _Float16 Bt[64*272];  // [t_local][c], stride 272 (32B aligned rows)
  int blk = blockIdx.x;
  int ntb = blk & 3;
  int i   = blk >> 2;
  int b = i / F_, f = i % F_;
  int tbase = ntb*64;
  int tid = threadIdx.x;

  // stage x[b,:,f,tbase:tbase+64] into LDS t-major as f16 (coalesced global reads over t)
  int tl = tid & 63, cr = tid >> 6;
  bool tvalid = (tbase + tl) < T_;
  const float* xb = x + ((size_t)b*C_*F_ + f)*T_ + tbase + tl;
  for (int c = cr; c < C_; c += 4) {
    float v = tvalid ? xb[(size_t)c*F_*T_] : 0.f;
    Bt[tl*272 + c] = (_Float16)v;
  }
  __syncthreads();

  int wave = tid >> 5, lane = tid & 31;
  int lhalf = lane >> 4, lm = lane & 15;
  v8f zero8 = {0.f,0.f,0.f,0.f,0.f,0.f,0.f,0.f};
  v8f acc[2][4];
  #pragma unroll
  for (int mi = 0; mi < 2; ++mi)
    #pragma unroll
    for (int ni = 0; ni < 4; ++ni) acc[mi][ni] = zero8;

  int rowA0 = wave*32 + lm;  // rows of wv for m-tile 0 of this wave; m-tile 1 is +16
  #pragma unroll
  for (int kk2 = 0; kk2 < 8; ++kk2) {
    int kb = kk2*32 + lhalf*16;     // per-lane K base (A/B frags hold 16 contiguous K)
    v16h a0 = *(const v16h*)(wvh + (size_t)rowA0*C_ + kb);
    v16h a1 = *(const v16h*)(wvh + (size_t)(rowA0+16)*C_ + kb);
    v16h bb[4];
    #pragma unroll
    for (int ni = 0; ni < 4; ++ni)
      bb[ni] = *(const v16h*)(&Bt[(ni*16 + lm)*272 + kb]);
    #pragma unroll
    for (int ni = 0; ni < 4; ++ni) {
      acc[0][ni] = __builtin_amdgcn_wmma_f32_16x16x32_f16(false, a0, false, bb[ni],
                                                          (short)0, acc[0][ni], false, false);
      acc[1][ni] = __builtin_amdgcn_wmma_f32_16x16x32_f16(false, a1, false, bb[ni],
                                                          (short)0, acc[1][ni], false, false);
    }
  }

  // epilogue: out = (acc*sv + beta) * (dc[h]+df[t]+dt[f]) + x
  float dtf = dt[f];
  #pragma unroll
  for (int mi = 0; mi < 2; ++mi) {
    int hbase = wave*32 + mi*16 + lhalf*8;  // C/D layout: lanes 16-31 hold M+8
    #pragma unroll
    for (int ni = 0; ni < 4; ++ni) {
      int tloc = ni*16 + lm;                // C/D layout: N = lane&15
      int t = tbase + tloc;
      if (t < T_) {
        float dfv = df[t];
        #pragma unroll
        for (int j = 0; j < 8; ++j) {
          int h = hbase + j;
          float vv = acc[mi][ni][j]*sv[h] + beta[h];
          float scale = dc[h] + dfv + dtf;
          float res = (float)Bt[tloc*272 + h];
          out[((size_t)(b*C_ + h)*F_ + f)*T_ + t] = vv*scale + res;
        }
      }
    }
  }
}

extern "C" void kernel_launch(void* const* d_in, const int* in_sizes, int n_in,
                              void* d_out, int out_size, void* d_ws, size_t ws_size,
                              hipStream_t stream) {
  const float* x      = (const float*)d_in[0];
  const float* wq_c   = (const float*)d_in[1];
  const float* wk_c   = (const float*)d_in[2];
  const float* wq_f   = (const float*)d_in[3];
  const float* wk_f   = (const float*)d_in[4];
  const float* wq_t   = (const float*)d_in[5];
  const float* wk_t   = (const float*)d_in[6];
  const float* wv     = (const float*)d_in[7];
  const float* bv     = (const float*)d_in[8];
  const float* gq_c   = (const float*)d_in[9];
  const float* bq_c   = (const float*)d_in[10];
  const float* gk_c   = (const float*)d_in[11];
  const float* bk_c   = (const float*)d_in[12];
  const float* gq_f   = (const float*)d_in[13];
  const float* bq_f   = (const float*)d_in[14];
  const float* gk_f   = (const float*)d_in[15];
  const float* bk_f   = (const float*)d_in[16];
  const float* gq_t   = (const float*)d_in[17];
  const float* bq_t   = (const float*)d_in[18];
  const float* gk_t   = (const float*)d_in[19];
  const float* bk_t   = (const float*)d_in[20];
  const float* glog_c = (const float*)d_in[21];
  const float* glog_f = (const float*)d_in[23];
  const float* glog_t = (const float*)d_in[25];
  const float* gv     = (const float*)d_in[27];
  const float* bbv    = (const float*)d_in[28];
  float* out = (float*)d_out;

  char* ws = (char*)d_ws;
  float*    aq_c = (float*)(ws + O_AQC);
  float*    ak_c = (float*)(ws + O_AKC);
  float*    aq_f = (float*)(ws + O_AQF);
  float*    ak_f = (float*)(ws + O_AKF);
  float*    aq_t = (float*)(ws + O_AQT);
  float*    ak_t = (float*)(ws + O_AKT);
  float*    cons = (float*)(ws + O_CONS);
  float*    sv   = (float*)(ws + O_SV);
  float*    beta = (float*)(ws + O_BETA);
  float*    dc   = (float*)(ws + O_DC);
  float*    df   = (float*)(ws + O_DF);
  float*    dt   = (float*)(ws + O_DT);
  _Float16* wvh  = (_Float16*)(ws + O_WVH);
  float*    qp_c = (float*)(ws + O_QPC);
  float*    kp_c = (float*)(ws + O_KPC);
  float*    qp_f = (float*)(ws + O_QPF);
  float*    kp_f = (float*)(ws + O_KPF);
  float*    qp_t = (float*)(ws + O_QPT);
  float*    kp_t = (float*)(ws + O_KPT);
  float*    cc   = (float*)(ws + O_CC);
  float*    cf   = (float*)(ws + O_CF);
  float*    pt   = (float*)(ws + O_PT);

  k0a<<<1, 256, 0, stream>>>(wq_c, wk_c, gq_c, gk_c, aq_c, ak_c);
  k0b<<<1, 256, 0, stream>>>(wq_f, wk_f, wq_t, wk_t, gq_f, gk_f, gq_t, gk_t,
                             aq_f, ak_f, aq_t, ak_t);
  k0c<<<256, 256, 0, stream>>>(wv, wvh);
  k0d<<<1, 256, 0, stream>>>(bq_c, bk_c, bq_f, bk_f, bq_t, bk_t,
                             glog_c, glog_f, glog_t, gv, bv, bbv, cons, sv, beta);

  k1a<<<NI*C_/8, 256, 0, stream>>>(x, aq_c, ak_c, cons, qp_c, kp_c);
  k1b<<<NI, 256, 0, stream>>>(x, aq_f, ak_f, aq_t, ak_t, cons, qp_f, kp_f, qp_t, kp_t);

  k2<<<NI, 256, 0, stream>>>(qp_c, kp_c, cons, cc);
  k3<<<NI, 256, 0, stream>>>(qp_f, kp_f, cons, cf);
  k4<<<NB4, 256, 0, stream>>>(qp_t, kp_t, cons, pt);
  kred<<<1, 256, 0, stream>>>(cc, cf, pt, dc, df, dt);

  k5<<<NI*4, 256, 0, stream>>>(x, wvh, sv, beta, dc, df, dt, out);
}
